// AttentionBlock_36867999269067
// MI455X (gfx1250) — compile-verified, matrix operand order checked by reading, not run on hardware
//
#include <hip/hip_runtime.h>
#include <hip/hip_bf16.h>

typedef __bf16 bf16;
typedef __attribute__((ext_vector_type(16))) __bf16 v16bf;
typedef __attribute__((ext_vector_type(8)))  __bf16 v8bf;
typedef __attribute__((ext_vector_type(8)))  float  v8f;

#define DEV static __device__ __forceinline__

constexpr int B_  = 8;
constexpr int C_  = 512;
constexpr int S_  = 1024;   // 32*32 tokens
constexpr int NH_ = 8;
constexpr int HD_ = 64;
constexpr int NG_ = 32;
constexpr int CG_ = 16;     // C/NG
constexpr float EPS_ = 1e-6f;

// ---------------------------------------------------------------------------
// WMMA fragment helpers (CDNA5 16x16x32 bf16, ISA 7.12.2 layouts)
// ---------------------------------------------------------------------------
DEV v16bf load_frag(const bf16* __restrict__ p0, int ld, int lane) {
  const bf16* p = p0 + (size_t)(lane & 15) * (size_t)ld + ((lane >> 4) << 3);
  v8bf lo = *(const v8bf*)p;
  v8bf hi = *(const v8bf*)(p + 16);
  return __builtin_shufflevector(lo, hi, 0,1,2,3,4,5,6,7,8,9,10,11,12,13,14,15);
}

DEV v8f wmma_bf16(v16bf a, v16bf b, v8f c) {
  return __builtin_amdgcn_wmma_f32_16x16x32_bf16(false, a, false, b, (short)0, c,
                                                 false, false);
}

DEV float silu(float x) { return x / (1.f + __expf(-x)); }

// DPP ROW_XMASK lane-swap within a 16-lane row (pure VALU, no LDS round-trip)
template<int MASK>
DEV float row_xmask(float v) {
  int r = __builtin_amdgcn_update_dpp(0, __float_as_int(v), 0x160 | MASK,
                                      0xf, 0xf, true);
  return __int_as_float(r);
}

DEV float row_max16(float v) {
  v = fmaxf(v, row_xmask<1>(v));
  v = fmaxf(v, row_xmask<2>(v));
  v = fmaxf(v, row_xmask<4>(v));
  v = fmaxf(v, row_xmask<8>(v));
  return v;
}

// ---------------------------------------------------------------------------
// WMMA GEMM:  out[M,N] = A[M,K](bf16) * W[N,K]^T(bf16)  (+bias)(+SiLU)
// One wave per 16x64 strip (A fragment reused across 4 WMMAs).
// TRANSP stores out as (B,N,S) channel-major.
// ---------------------------------------------------------------------------
template<bool SILU, bool TRANSP, typename OutT>
__global__ void gemm16(const bf16* __restrict__ A, int lda,
                       const bf16* __restrict__ W, int ldw,
                       const float* __restrict__ bias,
                       OutT* __restrict__ out, int ldo,
                       int M, int N, int K) {
  int wave = (blockIdx.x * blockDim.x + threadIdx.x) >> 5;
  int lane = threadIdx.x & 31;
  int ntiles = N >> 6;                      // 64-column strips
  int total  = (M >> 4) * ntiles;
  if (wave >= total) return;
  int mt = wave / ntiles, nt = wave % ntiles;
  const bf16* a0 = A + (size_t)(mt * 16) * (size_t)lda;
  const bf16* w0 = W + (size_t)(nt * 64) * (size_t)ldw;
  v8f acc[4] = {};
  for (int k = 0; k < K; k += 32) {
    v16bf af = load_frag(a0 + k, lda, lane);
#pragma unroll
    for (int t = 0; t < 4; t++) {
      v16bf wf = load_frag(w0 + (size_t)(t * 16) * (size_t)ldw + k, ldw, lane);
      acc[t] = wmma_bf16(af, wf, acc[t]);
    }
  }
  int col = lane & 15;
  int rb  = (lane >> 4) << 3;
#pragma unroll
  for (int t = 0; t < 4; t++) {
    int n = nt * 64 + t * 16 + col;
    float bv = bias ? bias[n] : 0.f;
#pragma unroll
    for (int j = 0; j < 8; j++) {
      float v = acc[t][j] + bv;
      if (SILU) v = silu(v);
      int m = mt * 16 + rb + j;
      if (TRANSP) {             // out[(b*N + n)*S + s], m = b*S + s
        int b = m >> 10, s = m & (S_ - 1);
        out[((size_t)(b * N + n)) * S_ + s] = (OutT)v;
      } else {
        out[(size_t)m * (size_t)ldo + n] = (OutT)v;
      }
    }
  }
}

// ---------------------------------------------------------------------------
// Flash attention core: one wave owns 16 query rows, streams keys 32 at a time.
// Row max:  DPP row_xmask VALU reduction (16-lane rows match C-layout halves).
// Row sum:  extra WMMA against an all-ones B fragment (5th accumulator tile),
//           rescaled by alpha like the output tiles -> no cross-lane sum at all.
// P tile goes through LDS (C-layout -> A-layout), ordered by s_wait_dscnt.
// ---------------------------------------------------------------------------
template<int DQK>
DEV void flash_core(const bf16* __restrict__ Qrow, int ldq,
                    const bf16* __restrict__ K0, int ldk,
                    const bf16* __restrict__ Vt,
                    bf16* __restrict__ OutRow, int ldo,
                    float scale, bf16* lds, int lane) {
  constexpr int KC = DQK / 32;
  v16bf qa[KC];
#pragma unroll
  for (int kc = 0; kc < KC; kc++) qa[kc] = load_frag(Qrow + kc * 32, ldq, lane);

  v16bf ones;
#pragma unroll
  for (int i = 0; i < 16; i++) ones[i] = (bf16)1.f;

  v8f o0 = {}, o1 = {}, o2 = {}, o3 = {}, ol = {};
  float mrow[8];
#pragma unroll
  for (int j = 0; j < 8; j++) mrow[j] = -1e30f;

  int col = lane & 15;
  int rb  = (lane >> 4) << 3;

  for (int kb = 0; kb < S_; kb += 32) {
    v8f s0 = {}, s1 = {};
#pragma unroll
    for (int kc = 0; kc < KC; kc++) {
      v16bf b0 = load_frag(K0 + (size_t)kb * ldk + kc * 32, ldk, lane);
      v16bf b1 = load_frag(K0 + (size_t)(kb + 16) * ldk + kc * 32, ldk, lane);
      s0 = wmma_bf16(qa[kc], b0, s0);
      s1 = wmma_bf16(qa[kc], b1, s1);
    }
#pragma unroll
    for (int j = 0; j < 8; j++) {
      float a = s0[j] * scale, b = s1[j] * scale;
      float mx = row_max16(fmaxf(a, b));
      float mn = fmaxf(mrow[j], mx);
      float al = __expf(mrow[j] - mn);
      mrow[j] = mn;
      float p0 = __expf(a - mn), p1 = __expf(b - mn);
      o0[j] *= al; o1[j] *= al; o2[j] *= al; o3[j] *= al; ol[j] *= al;
      lds[(rb + j) * 32 + col]      = (bf16)p0;
      lds[(rb + j) * 32 + 16 + col] = (bf16)p1;
    }
    asm volatile("s_wait_dscnt 0" ::: "memory");   // CDNA5 split DS counter
    v16bf pa = load_frag(lds, 32, lane);           // P as A-fragment (K = keys)
    o0 = wmma_bf16(pa, load_frag(Vt + kb,           S_, lane), o0);
    o1 = wmma_bf16(pa, load_frag(Vt + 16 * S_ + kb, S_, lane), o1);
    o2 = wmma_bf16(pa, load_frag(Vt + 32 * S_ + kb, S_, lane), o2);
    o3 = wmma_bf16(pa, load_frag(Vt + 48 * S_ + kb, S_, lane), o3);
    ol = wmma_bf16(pa, ones, ol);                  // row sums, no shuffles
  }
#pragma unroll
  for (int j = 0; j < 8; j++) {
    float inv = 1.f / ol[j];
    bf16* po = OutRow + (size_t)(rb + j) * (size_t)ldo;
    po[col]      = (bf16)(o0[j] * inv);
    po[16 + col] = (bf16)(o1[j] * inv);
    po[32 + col] = (bf16)(o2[j] * inv);
    po[48 + col] = (bf16)(o3[j] * inv);
  }
}

// standard 8-head attention: 4 waves/block, wave = (b, h, 16-row tile)
__global__ void mhsa_kernel(const bf16* __restrict__ q, const bf16* __restrict__ k,
                            const bf16* __restrict__ vT, bf16* __restrict__ comb) {
  __shared__ bf16 lds[4][16 * 32];
  int wid = threadIdx.x >> 5, lane = threadIdx.x & 31;
  int gw = blockIdx.x * 4 + wid;            // 0 .. B*NH*(S/16)-1 = 4095
  int b  = gw >> 9;
  int h  = (gw >> 6) & 7;
  int mt = gw & 63;
  flash_core<64>(q + ((size_t)(b * S_ + mt * 16)) * C_ + h * HD_, C_,
                 k + (size_t)b * S_ * C_ + h * HD_, C_,
                 vT + ((size_t)(b * C_ + h * HD_)) * S_,
                 comb + ((size_t)(b * S_ + mt * 16)) * (2 * C_) + h * HD_, 2 * C_,
                 0.125f, lds[wid], lane);
}

// multi-scale single-head branch (d=256): block = one 16-row tile; each of the
// 4 waves recomputes the (cheap) scores and produces 64 of the 256 out columns.
__global__ void ms_kernel(const bf16* __restrict__ qm, const bf16* __restrict__ km,
                          const bf16* __restrict__ vmT, bf16* __restrict__ comb,
                          int outbase) {
  __shared__ bf16 lds[4][16 * 32];
  int wid = threadIdx.x >> 5, lane = threadIdx.x & 31;
  int b = blockIdx.x >> 6, mt = blockIdx.x & 63;
  flash_core<256>(qm + ((size_t)(b * S_ + mt * 16)) * (C_ / 2), C_ / 2,
                  km + (size_t)b * S_ * (C_ / 2), C_ / 2,
                  vmT + ((size_t)(b * (C_ / 2)) + wid * 64) * S_,
                  comb + ((size_t)(b * S_ + mt * 16)) * (2 * C_) + outbase + wid * 64,
                  2 * C_, 0.0625f, lds[wid], lane);
}

// ---------------------------------------------------------------------------
// GroupNorm stats (channel-major or token-major via strides)
// ---------------------------------------------------------------------------
__global__ void gn_stats(const float* __restrict__ x, float* __restrict__ stats,
                         int strideC, int strideS) {
  int b = blockIdx.x >> 5, g = blockIdx.x & 31;
  const float* basep = x + (size_t)b * C_ * S_;
  float s = 0.f, ss = 0.f;
  for (int i = threadIdx.x; i < CG_ * S_; i += 256) {
    int c  = g * CG_ + (i >> 10);
    int sp = i & (S_ - 1);
    float v = basep[(size_t)c * strideC + (size_t)sp * strideS];
    s += v; ss += v * v;
  }
  __shared__ float sh1[256], sh2[256];
  sh1[threadIdx.x] = s; sh2[threadIdx.x] = ss;
  __syncthreads();
  for (int st = 128; st > 0; st >>= 1) {
    if (threadIdx.x < st) { sh1[threadIdx.x] += sh1[threadIdx.x + st];
                            sh2[threadIdx.x] += sh2[threadIdx.x + st]; }
    __syncthreads();
  }
  if (threadIdx.x == 0) {
    float mean = sh1[0] / (CG_ * S_);
    float var  = sh2[0] / (CG_ * S_) - mean * mean;
    stats[blockIdx.x * 2]     = mean;
    stats[blockIdx.x * 2 + 1] = rsqrtf(var + EPS_);
  }
}

__global__ void apply_pre(const float* __restrict__ x, const float* __restrict__ st,
                          const float* __restrict__ g, const float* __restrict__ be,
                          const float* __restrict__ pos, float* __restrict__ hs) {
  int idx = blockIdx.x * blockDim.x + threadIdx.x;  // (b,c,s)
  if (idx >= B_ * C_ * S_) return;
  int s = idx & (S_ - 1);
  int c = (idx >> 10) & (C_ - 1);
  int b = idx >> 19;
  int grp = c >> 4;
  float mu = st[(b * NG_ + grp) * 2], rs = st[(b * NG_ + grp) * 2 + 1];
  hs[idx] = (x[idx] - mu) * rs * g[c] + be[c] + pos[c * S_ + s];
}

// spatial gating convs (~1.5% of total FLOPs -> plain VALU)
__global__ void conv1x1_silu(const float* __restrict__ hs, const float* __restrict__ w,
                             const float* __restrict__ bias, float* __restrict__ out) {
  int idx = blockIdx.x * blockDim.x + threadIdx.x;  // (b,co,s), co<64
  if (idx >= B_ * 64 * S_) return;
  int s = idx & (S_ - 1);
  int co = (idx >> 10) & 63;
  int b = idx >> 16;
  const float* xin = hs + (size_t)b * C_ * S_ + s;
  const float* wr  = w + co * C_;
  float acc = bias[co];
  for (int ci = 0; ci < C_; ci++) acc += wr[ci] * xin[(size_t)ci * S_];
  out[idx] = silu(acc);
}

__global__ void conv3x3_silu(const float* __restrict__ a1, const float* __restrict__ w,
                             const float* __restrict__ bias, float* __restrict__ out) {
  int idx = blockIdx.x * blockDim.x + threadIdx.x;  // (b,co,s)
  if (idx >= B_ * 64 * S_) return;
  int s = idx & (S_ - 1);
  int x = s & 31, y = s >> 5;
  int co = (idx >> 10) & 63;
  int b = idx >> 16;
  float acc = bias[co];
  for (int ci = 0; ci < 64; ci++) {
    const float* xin = a1 + ((size_t)b * 64 + ci) * S_;
    const float* wr  = w + ((size_t)co * 64 + ci) * 9;
#pragma unroll
    for (int t = 0; t < 9; t++) {
      int yy = y + t / 3 - 1, xx = x + t % 3 - 1;
      if ((unsigned)yy < 32u && (unsigned)xx < 32u) acc += wr[t] * xin[yy * 32 + xx];
    }
  }
  out[idx] = silu(acc);
}

__global__ void conv_gate(const float* __restrict__ a2, const float* __restrict__ w,
                          const float* __restrict__ bias, float* __restrict__ sw) {
  int idx = blockIdx.x * blockDim.x + threadIdx.x;  // (b,s)
  if (idx >= B_ * S_) return;
  int s = idx & (S_ - 1);
  int b = idx >> 10;
  float acc = bias[0];
  for (int ci = 0; ci < 64; ci++) acc += w[ci] * a2[((size_t)b * 64 + ci) * S_ + s];
  sw[idx] = 1.f / (1.f + __expf(-acc));
}

// mid GN -> token-major bf16 sh; spatial_enhanced -> token-major bf16 msh
__global__ void build_tokens(const float* __restrict__ hs, const float* __restrict__ st,
                             const float* __restrict__ g, const float* __restrict__ be,
                             const float* __restrict__ sw,
                             bf16* __restrict__ shT, bf16* __restrict__ mshT) {
  int idx = blockIdx.x * blockDim.x + threadIdx.x;  // (b,s,c)
  if (idx >= B_ * S_ * C_) return;
  int c = idx & (C_ - 1);
  int s = (idx >> 9) & (S_ - 1);
  int b = idx >> 19;
  int grp = c >> 4;
  float v  = hs[((size_t)b * C_ + c) * S_ + s];
  float mu = st[(b * NG_ + grp) * 2], rs = st[(b * NG_ + grp) * 2 + 1];
  shT[idx]  = (bf16)((v - mu) * rs * g[c] + be[c]);
  mshT[idx] = (bf16)(v * sw[b * S_ + s]);
}

__global__ void apply_post(const float* __restrict__ finalT, const float* __restrict__ st,
                           const float* __restrict__ g, const float* __restrict__ be,
                           const float* __restrict__ residual, float* __restrict__ out) {
  int idx = blockIdx.x * blockDim.x + threadIdx.x;  // (b,c,s)
  if (idx >= B_ * C_ * S_) return;
  int s = idx & (S_ - 1);
  int c = (idx >> 10) & (C_ - 1);
  int b = idx >> 19;
  int grp = c >> 4;
  float v  = finalT[((size_t)b * S_ + s) * C_ + c];
  float mu = st[(b * NG_ + grp) * 2], rs = st[(b * NG_ + grp) * 2 + 1];
  out[idx] = (v - mu) * rs * g[c] + be[c] + residual[idx];
}

__global__ void f32_to_bf16(const float* __restrict__ in, bf16* __restrict__ out, int n) {
  int i = blockIdx.x * blockDim.x + threadIdx.x;
  if (i < n) out[i] = (bf16)in[i];
}

// ---------------------------------------------------------------------------
extern "C" void kernel_launch(void* const* d_in, const int* in_sizes, int n_in,
                              void* d_out, int out_size, void* d_ws, size_t ws_size,
                              hipStream_t stream) {
  const float* x      = (const float*)d_in[0];
  const float* pre_g  = (const float*)d_in[1];
  const float* pre_b  = (const float*)d_in[2];
  const float* norm_g = (const float*)d_in[3];
  const float* norm_b = (const float*)d_in[4];
  const float* post_g = (const float*)d_in[5];
  const float* post_b = (const float*)d_in[6];
  const float* pos    = (const float*)d_in[7];
  const float* sa_w1  = (const float*)d_in[8];
  const float* sa_b1  = (const float*)d_in[9];
  const float* sa_w2  = (const float*)d_in[10];
  const float* sa_b2  = (const float*)d_in[11];
  const float* sa_w3  = (const float*)d_in[12];
  const float* sa_b3  = (const float*)d_in[13];
  const float* wq = (const float*)d_in[14];
  const float* wk = (const float*)d_in[15];
  const float* wv = (const float*)d_in[16];
  const float* wm[6] = { (const float*)d_in[17], (const float*)d_in[18],
                         (const float*)d_in[19], (const float*)d_in[20],
                         (const float*)d_in[21], (const float*)d_in[22] };
  const float* ff_w1 = (const float*)d_in[23];
  const float* ff_b1 = (const float*)d_in[24];
  const float* ff_w2 = (const float*)d_in[25];
  const float* ff_b2 = (const float*)d_in[26];
  const float* out_w = (const float*)d_in[27];
  const float* out_b = (const float*)d_in[28];
  float* out = (float*)d_out;
  (void)in_sizes; (void)n_in; (void)out_size; (void)ws_size;

  unsigned char* base = (unsigned char*)d_ws;
  size_t off = 0;
  auto alloc = [&](size_t bytes) -> void* {
    void* p = base + off;
    off = (off + bytes + 255) & ~(size_t)255;
    return p;
  };

  float* hs     = (float*)alloc((size_t)B_ * C_ * S_ * 4);   // reused as finalT
  float* finalT = hs;
  float* a1     = (float*)alloc((size_t)B_ * 64 * S_ * 4);
  float* a2     = (float*)alloc((size_t)B_ * 64 * S_ * 4);
  float* sw     = (float*)alloc((size_t)B_ * S_ * 4);
  float* st_pre  = (float*)alloc(B_ * NG_ * 2 * 4);
  float* st_mid  = (float*)alloc(B_ * NG_ * 2 * 4);
  float* st_post = (float*)alloc(B_ * NG_ * 2 * 4);
  bf16* shT   = (bf16*)alloc((size_t)B_ * S_ * C_ * 2);
  bf16* mshT  = (bf16*)alloc((size_t)B_ * S_ * C_ * 2);
  bf16* wq_bf = (bf16*)alloc((size_t)C_ * C_ * 2);
  bf16* wk_bf = (bf16*)alloc((size_t)C_ * C_ * 2);
  bf16* wv_bf = (bf16*)alloc((size_t)C_ * C_ * 2);
  bf16* wm_bf[6];
  for (int i = 0; i < 6; i++) wm_bf[i] = (bf16*)alloc((size_t)(C_ / 2) * C_ * 2);
  bf16* ffw1_bf = (bf16*)alloc((size_t)C_ * 2 * C_ * 2);
  bf16* ffw2_bf = (bf16*)alloc((size_t)C_ * C_ * 2);
  bf16* outw_bf = (bf16*)alloc((size_t)C_ * C_ * 2);
  bf16* qb   = (bf16*)alloc((size_t)B_ * S_ * C_ * 2);       // reused as ff1
  bf16* kb   = (bf16*)alloc((size_t)B_ * S_ * C_ * 2);       // reused as ff2
  bf16* vTb  = (bf16*)alloc((size_t)B_ * C_ * S_ * 2);
  bf16* qm   = (bf16*)alloc((size_t)B_ * S_ * (C_ / 2) * 2);
  bf16* km   = (bf16*)alloc((size_t)B_ * S_ * (C_ / 2) * 2);
  bf16* vmT  = (bf16*)alloc((size_t)B_ * (C_ / 2) * S_ * 2);
  bf16* comb = (bf16*)alloc((size_t)B_ * S_ * 2 * C_ * 2);
  bf16* ff1 = qb;
  bf16* ff2 = kb;

  auto cvt = [&](const float* s, bf16* d, int n) {
    f32_to_bf16<<<(n + 255) / 256, 256, 0, stream>>>(s, d, n);
  };
  cvt(wq, wq_bf, C_ * C_); cvt(wk, wk_bf, C_ * C_); cvt(wv, wv_bf, C_ * C_);
  for (int i = 0; i < 6; i++) cvt(wm[i], wm_bf[i], (C_ / 2) * C_);
  cvt(ff_w1, ffw1_bf, C_ * 2 * C_);
  cvt(ff_w2, ffw2_bf, C_ * C_);
  cvt(out_w, outw_bf, C_ * C_);

  // pre-GN + positional embedding
  gn_stats<<<B_ * NG_, 256, 0, stream>>>(x, st_pre, S_, 1);
  apply_pre<<<(B_ * C_ * S_) / 256, 256, 0, stream>>>(x, st_pre, pre_g, pre_b, pos, hs);

  // spatial gate
  conv1x1_silu<<<(B_ * 64 * S_) / 256, 256, 0, stream>>>(hs, sa_w1, sa_b1, a1);
  conv3x3_silu<<<(B_ * 64 * S_) / 256, 256, 0, stream>>>(a1, sa_w2, sa_b2, a2);
  conv_gate<<<(B_ * S_) / 256, 256, 0, stream>>>(a2, sa_w3, sa_b3, sw);

  // mid-GN -> token-major bf16; spatial_enhanced -> token-major bf16
  gn_stats<<<B_ * NG_, 256, 0, stream>>>(hs, st_mid, S_, 1);
  build_tokens<<<(B_ * S_ * C_) / 256, 256, 0, stream>>>(hs, st_mid, norm_g, norm_b,
                                                         sw, shT, mshT);

  const int M = B_ * S_;
  auto blocks = [](int m, int n) { return ((m / 16) * (n / 64) + 3) / 4; };

  // QKV projections (V stored transposed (B,C,S) for the PV wmma B-fragments)
  gemm16<false, false, bf16><<<blocks(M, C_), 128, 0, stream>>>(
      shT, C_, wq_bf, C_, nullptr, qb, C_, M, C_, C_);
  gemm16<false, false, bf16><<<blocks(M, C_), 128, 0, stream>>>(
      shT, C_, wk_bf, C_, nullptr, kb, C_, M, C_, C_);
  gemm16<false, true, bf16><<<blocks(M, C_), 128, 0, stream>>>(
      shT, C_, wv_bf, C_, nullptr, vTb, C_, M, C_, C_);

  mhsa_kernel<<<(B_ * NH_ * (S_ / 16)) / 4, 128, 0, stream>>>(qb, kb, vTb, comb);

  // two multi-scale branches, sequential buffer reuse
  for (int br = 0; br < 2; br++) {
    gemm16<false, false, bf16><<<blocks(M, C_ / 2), 128, 0, stream>>>(
        mshT, C_, wm_bf[br * 3 + 0], C_, nullptr, qm, C_ / 2, M, C_ / 2, C_);
    gemm16<false, false, bf16><<<blocks(M, C_ / 2), 128, 0, stream>>>(
        shT, C_, wm_bf[br * 3 + 1], C_, nullptr, km, C_ / 2, M, C_ / 2, C_);
    gemm16<false, true, bf16><<<blocks(M, C_ / 2), 128, 0, stream>>>(
        shT, C_, wm_bf[br * 3 + 2], C_, nullptr, vmT, C_ / 2, M, C_ / 2, C_);
    ms_kernel<<<B_ * (S_ / 16), 128, 0, stream>>>(qm, km, vmT, comb,
                                                  C_ + br * (C_ / 2));
  }

  // fusion MLP + output projection (final in fp32 for post-GN)
  gemm16<true, false, bf16><<<blocks(M, C_), 128, 0, stream>>>(
      comb, 2 * C_, ffw1_bf, 2 * C_, ff_b1, ff1, C_, M, C_, 2 * C_);
  gemm16<false, false, bf16><<<blocks(M, C_), 128, 0, stream>>>(
      ff1, C_, ffw2_bf, C_, ff_b2, ff2, C_, M, C_, C_);
  gemm16<false, false, float><<<blocks(M, C_), 128, 0, stream>>>(
      ff2, C_, outw_bf, C_, out_b, finalT, C_, M, C_, C_);

  // post-GN (+ residual) back to (B,C,H,W)
  gn_stats<<<B_ * NG_, 256, 0, stream>>>(finalT, st_post, 1, C_);
  apply_post<<<(B_ * C_ * S_) / 256, 256, 0, stream>>>(finalT, st_post, post_g,
                                                       post_b, x, out);
}
